// SSMDiagScan_84731114816205
// MI455X (gfx1250) — matrix operand order repaired and checked by reading, not compiled, not run on hardware
//
#include <hip/hip_runtime.h>
#include <hip/hip_bf16.h>

// ---------------------------------------------------------------------------
// Problem constants (match reference)
// ---------------------------------------------------------------------------
#define BB 8
#define TT 2048
#define DD 2048
#define RR 4
#define BD (BB * DD)          // 16384 channels
#define BD_SHIFT 14

// T-chunking for the scan/mean passes (parallelism: BD*NCHUNK threads)
#define NCHUNK 32
#define LCH (TT / NCHUNK)     // 64 = 2^6

// GEMM tiling
#define TILE_M 128
#define TILE_N 128
#define KT 64
#define LDS_PAD (KT + 8)      // halves per row in LDS (16B pad: dodges bank conflicts)

typedef __attribute__((ext_vector_type(16))) __bf16 v16bf;
typedef __attribute__((ext_vector_type(8)))  float  v8f;

union FragBF16 {
    v16bf v;
    uint4 q[2];
};

static __device__ __forceinline__ unsigned short f32_to_bf16_bits(float f) {
    unsigned u = __builtin_bit_cast(unsigned, f);
    u += 0x7FFFu + ((u >> 16) & 1u);      // round-to-nearest-even
    return (unsigned short)(u >> 16);
}

// A fragment (16x32 bf16): lanes 0-15 -> M=lane, K in {0..7, 16..23};
// lanes 16-31 -> M=lane-16, K in {8..15, 24..31}. Two b128 LDS loads.
static __device__ __forceinline__ v16bf load_frag_a(const __bf16* lds, int m_off,
                                                    int k_off, int lane) {
    int half = (lane >> 4) & 1;
    int r    = lane & 15;
    const __bf16* p = lds + (m_off + r) * LDS_PAD + k_off + half * 8;
    FragBF16 f;
    f.q[0] = *reinterpret_cast<const uint4*>(p);
    f.q[1] = *reinterpret_cast<const uint4*>(p + 16);
    return f.v;
}

// B fragment (32x16 bf16): lanes 0-15 -> N=lane, K in {0..7, 16..23};
// lanes 16-31 -> N=lane-16, K in {8..15, 24..31}. LDS holds B as [n][k] (gate_w native).
static __device__ __forceinline__ v16bf load_frag_b(const __bf16* lds, int n_off,
                                                    int k_off, int lane) {
    int half = (lane >> 4) & 1;
    int c    = lane & 15;
    const __bf16* p = lds + (n_off + c) * LDS_PAD + k_off + half * 8;
    FragBF16 f;
    f.q[0] = *reinterpret_cast<const uint4*>(p);
    f.q[1] = *reinterpret_cast<const uint4*>(p + 16);
    return f.v;
}

// Low 32 bits of a flat "shared" pointer ARE the LDS byte address
// (LDS aperture lives entirely in ADDR[63:32] per CDNA5 ISA 10.2).
static __device__ __forceinline__ unsigned lds_addr_of(const void* p) {
    return (unsigned)(uintptr_t)p;
}

// Issue 4 async 16B global->LDS copies for one 128-row tile slab.
// Each thread covers rows {row, row+32, row+64, row+96}, one 16B chunk each.
static __device__ __forceinline__ void async_copy_tile(const __bf16* gbase,
                                                       __bf16* lds_base,
                                                       int row, int chunk) {
#pragma unroll
    for (int i = 0; i < 4; ++i) {
        int r = row + i * 32;
        unsigned dst = lds_addr_of(lds_base + r * LDS_PAD + chunk * 8);
        unsigned long long ga =
            (unsigned long long)(uintptr_t)(gbase + (size_t)r * DD + chunk * 8);
        asm volatile("global_load_async_to_lds_b128 %0, %1, off"
                     :: "v"(dst), "v"(ga) : "memory");
    }
}

// WMMA inner tile compute: all 6 fragments per k-slice loaded into distinct
// registers before the 8 WMMAs so the scheduler can hide DS latency.
static __device__ __forceinline__ void wmma_tile(const __bf16* lA, const __bf16* lB,
                                                 int wm, int wn, int lane,
                                                 v8f acc[4][2]) {
#pragma unroll
    for (int kk = 0; kk < KT; kk += 32) {
        v16bf bfr0 = load_frag_b(lB, wn +  0, kk, lane);
        v16bf bfr1 = load_frag_b(lB, wn + 16, kk, lane);
        v16bf af0  = load_frag_a(lA, wm +  0, kk, lane);
        v16bf af1  = load_frag_a(lA, wm + 16, kk, lane);
        v16bf af2  = load_frag_a(lA, wm + 32, kk, lane);
        v16bf af3  = load_frag_a(lA, wm + 48, kk, lane);
        acc[0][0] = __builtin_amdgcn_wmma_f32_16x16x32_bf16(
            false, af0, false, bfr0, (short)0, acc[0][0], false, false);
        acc[0][1] = __builtin_amdgcn_wmma_f32_16x16x32_bf16(
            false, af0, false, bfr1, (short)0, acc[0][1], false, false);
        acc[1][0] = __builtin_amdgcn_wmma_f32_16x16x32_bf16(
            false, af1, false, bfr0, (short)0, acc[1][0], false, false);
        acc[1][1] = __builtin_amdgcn_wmma_f32_16x16x32_bf16(
            false, af1, false, bfr1, (short)0, acc[1][1], false, false);
        acc[2][0] = __builtin_amdgcn_wmma_f32_16x16x32_bf16(
            false, af2, false, bfr0, (short)0, acc[2][0], false, false);
        acc[2][1] = __builtin_amdgcn_wmma_f32_16x16x32_bf16(
            false, af2, false, bfr1, (short)0, acc[2][1], false, false);
        acc[3][0] = __builtin_amdgcn_wmma_f32_16x16x32_bf16(
            false, af3, false, bfr0, (short)0, acc[3][0], false, false);
        acc[3][1] = __builtin_amdgcn_wmma_f32_16x16x32_bf16(
            false, af3, false, bfr1, (short)0, acc[3][1], false, false);
    }
}

// Shared epilogue: bias + sigmoid + u*g -> out (u_g).
// C/D layout: lanes 0-15 -> M=r, lanes 16-31 -> M=8+r; N = lane&15.
static __device__ __forceinline__ void gate_epilogue(const float* u,
                                                     const float* gate_b,
                                                     float* out, int row0, int col0,
                                                     int wm, int wn, int lane,
                                                     v8f acc[4][2]) {
    const int half = lane >> 4;
    const int cc   = lane & 15;
#pragma unroll
    for (int mf = 0; mf < 4; ++mf) {
        int rbase = row0 + wm + mf * 16 + half * 8;
#pragma unroll
        for (int nf = 0; nf < 2; ++nf) {
            int col  = col0 + wn + nf * 16 + cc;
            float bias = gate_b[col];
#pragma unroll
            for (int r = 0; r < 8; ++r) {
                size_t off = (size_t)(rbase + r) * DD + col;
                float uval = u[off];
                float x = acc[mf][nf][r] + bias;
                float g = 1.0f / (1.0f + __expf(-x));
                out[off] = uval * g;
            }
        }
    }
}

// ---------------------------------------------------------------------------
// Kernel 0: f32 -> bf16 bulk convert (grid-stride over float4 chunks)
// ---------------------------------------------------------------------------
__global__ __launch_bounds__(256)
void cvt_bf16_kernel(const float* __restrict__ src, __bf16* __restrict__ dst,
                     int n4) {
    int i = blockIdx.x * blockDim.x + threadIdx.x;
    int stride = gridDim.x * blockDim.x;
    for (; i < n4; i += stride) {
        float4 v = reinterpret_cast<const float4*>(src)[i];
        unsigned p0 = ((unsigned)f32_to_bf16_bits(v.y) << 16) | f32_to_bf16_bits(v.x);
        unsigned p1 = ((unsigned)f32_to_bf16_bits(v.w) << 16) | f32_to_bf16_bits(v.z);
        reinterpret_cast<uint2*>(dst)[i] = make_uint2(p0, p1);
    }
}

// ---------------------------------------------------------------------------
// Kernel 1 (primary): async double-buffered bf16 GEMM + gate epilogue.
//   g = sigmoid(u @ gate_w^T + gate_b); u_g = u * g -> out
// ---------------------------------------------------------------------------
__global__ __launch_bounds__(256)
void ssm_gate_gemm_async_kernel(const __bf16* __restrict__ u_bf,
                                const __bf16* __restrict__ w_bf,
                                const float* __restrict__ u,
                                const float* __restrict__ gate_b,
                                float* __restrict__ out) {
    __shared__ __bf16 ldsA[2][TILE_M * LDS_PAD];
    __shared__ __bf16 ldsB[2][TILE_N * LDS_PAD];

    const int tid  = threadIdx.x;
    const int lane = tid & 31;
    const int wave = tid >> 5;
    const int wm   = (wave >> 2) * 64;
    const int wn   = (wave & 3) * 32;

    const int n_tiles = DD / TILE_N;
    const int tile_n  = blockIdx.x % n_tiles;
    const int tile_m  = blockIdx.x / n_tiles;
    const int row0    = tile_m * TILE_M;
    const int col0    = tile_n * TILE_N;

    const int row   = tid >> 3;
    const int chunk = tid & 7;

    v8f acc[4][2];
#pragma unroll
    for (int i = 0; i < 4; ++i)
#pragma unroll
        for (int j = 0; j < 2; ++j)
            acc[i][j] = (v8f){0.f, 0.f, 0.f, 0.f, 0.f, 0.f, 0.f, 0.f};

    const __bf16* Ag = u_bf + (size_t)row0 * DD;
    const __bf16* Bg = w_bf + (size_t)col0 * DD;

    async_copy_tile(Ag, ldsA[0], row, chunk);
    async_copy_tile(Bg, ldsB[0], row, chunk);

    int buf = 0;
    for (int k0 = 0; k0 < DD; k0 += KT) {
        __syncthreads();
        if (k0 + KT < DD) {
            async_copy_tile(Ag + (k0 + KT), ldsA[buf ^ 1], row, chunk);
            async_copy_tile(Bg + (k0 + KT), ldsB[buf ^ 1], row, chunk);
            asm volatile("s_wait_asynccnt 0x8" ::: "memory");
        } else {
            asm volatile("s_wait_asynccnt 0x0" ::: "memory");
        }
        __syncthreads();

        wmma_tile(ldsA[buf], ldsB[buf], wm, wn, lane, acc);
        buf ^= 1;
    }

    gate_epilogue(u, gate_b, out, row0, col0, wm, wn, lane, acc);
}

// ---------------------------------------------------------------------------
// Kernel 1 (fallback when ws too small): convert-in-kernel GEMM.
// ---------------------------------------------------------------------------
__global__ __launch_bounds__(256)
void ssm_gate_gemm_kernel(const float* __restrict__ u,
                          const float* __restrict__ gate_w,
                          const float* __restrict__ gate_b,
                          float* __restrict__ out) {
    __shared__ __bf16 ldsA[TILE_M * LDS_PAD];
    __shared__ __bf16 ldsB[TILE_N * LDS_PAD];

    const int tid  = threadIdx.x;
    const int lane = tid & 31;
    const int wave = tid >> 5;
    const int wm   = (wave >> 2) * 64;
    const int wn   = (wave & 3) * 32;

    const int n_tiles = DD / TILE_N;
    const int tile_n  = blockIdx.x % n_tiles;
    const int tile_m  = blockIdx.x / n_tiles;
    const int row0    = tile_m * TILE_M;
    const int col0    = tile_n * TILE_N;

    const int kq = (tid & 15) * 4;
    const int rr = tid >> 4;

    v8f acc[4][2];
#pragma unroll
    for (int i = 0; i < 4; ++i)
#pragma unroll
        for (int j = 0; j < 2; ++j)
            acc[i][j] = (v8f){0.f, 0.f, 0.f, 0.f, 0.f, 0.f, 0.f, 0.f};

    const float* Ag = u      + (size_t)row0 * DD;
    const float* Bg = gate_w + (size_t)col0 * DD;

    for (int k0 = 0; k0 < DD; k0 += KT) {
        __syncthreads();
#pragma unroll
        for (int i = 0; i < 8; ++i) {
            int m = rr + i * 16;
            float4 v = *reinterpret_cast<const float4*>(Ag + (size_t)m * DD + k0 + kq);
            unsigned p0 = ((unsigned)f32_to_bf16_bits(v.y) << 16) | f32_to_bf16_bits(v.x);
            unsigned p1 = ((unsigned)f32_to_bf16_bits(v.w) << 16) | f32_to_bf16_bits(v.z);
            *reinterpret_cast<uint2*>(ldsA + m * LDS_PAD + kq) = make_uint2(p0, p1);
        }
#pragma unroll
        for (int i = 0; i < 8; ++i) {
            int n = rr + i * 16;
            float4 v = *reinterpret_cast<const float4*>(Bg + (size_t)n * DD + k0 + kq);
            unsigned p0 = ((unsigned)f32_to_bf16_bits(v.y) << 16) | f32_to_bf16_bits(v.x);
            unsigned p1 = ((unsigned)f32_to_bf16_bits(v.w) << 16) | f32_to_bf16_bits(v.z);
            *reinterpret_cast<uint2*>(ldsB + n * LDS_PAD + kq) = make_uint2(p0, p1);
        }
        if (k0 + KT < DD) {
            __builtin_prefetch(Ag + (size_t)rr * DD + k0 + KT + kq, 0, 1);
            __builtin_prefetch(Bg + (size_t)rr * DD + k0 + KT + kq, 0, 1);
        }
        __syncthreads();

        wmma_tile(ldsA, ldsB, wm, wn, lane, acc);
    }

    gate_epilogue(u, gate_b, out, row0, col0, wm, wn, lane, acc);
}

// ---------------------------------------------------------------------------
// Kernel 2: per-chunk decayed sums + plain sums (fused mean partial).
//   e[c][bd]   = sum_{i<L} aa^(L-1-i) * u_g[b, c*L+i, d]
//   sum[c][bd] = sum_{i<L} u_g[b, c*L+i, d]
// One thread per (chunk, b, d): BD*NCHUNK = 512K threads.
// ---------------------------------------------------------------------------
__global__ __launch_bounds__(256)
void ssm_chunk_reduce_kernel(const float* __restrict__ ug,
                             const float* __restrict__ av,
                             float* __restrict__ e_part,
                             float* __restrict__ sum_part) {
    int idx = blockIdx.x * blockDim.x + threadIdx.x;
    int bd  = idx & (BD - 1);
    int c   = idx >> BD_SHIFT;
    int dch = bd & (DD - 1);
    int b   = bd >> 11;

    float aa  = tanhf(av[dch]);
    float aa2 = aa * aa;
    float aa3 = aa2 * aa;
    float aa4 = aa2 * aa2;

    const float* p = ug + ((size_t)b * TT + (size_t)c * LCH) * DD + dch;
    float e = 0.f, sum = 0.f;
#pragma unroll 1
    for (int i = 0; i < LCH; i += 4) {
        float x0 = p[(size_t)(i + 0) * DD];
        float x1 = p[(size_t)(i + 1) * DD];
        float x2 = p[(size_t)(i + 2) * DD];
        float x3 = p[(size_t)(i + 3) * DD];
        // e <- aa^4*e + aa^3*x0 + aa^2*x1 + aa*x2 + x3  (4 steps at once)
        e = fmaf(aa4, e, fmaf(aa3, x0, fmaf(aa2, x1, fmaf(aa, x2, x3))));
        sum += (x0 + x1) + (x2 + x3);
    }
    e_part[c * BD + bd]   = e;
    sum_part[c * BD + bd] = sum;
}

// ---------------------------------------------------------------------------
// Kernel 3: inter-chunk prefix (init states) + u_mean.
//   init[0]=0; init[c] = aa^L * init[c-1] + b * e[c-1];  u_mean = sum_c / T
// ---------------------------------------------------------------------------
__global__ __launch_bounds__(256)
void ssm_combine_kernel(const float* __restrict__ e_part,
                        const float* __restrict__ sum_part,
                        const float* __restrict__ av,
                        const float* __restrict__ bv,
                        float* __restrict__ init,
                        float* __restrict__ u_mean) {
    int bd  = blockIdx.x * blockDim.x + threadIdx.x;   // 0..BD-1
    int dch = bd & (DD - 1);
    float aa  = tanhf(av[dch]);
    float bb  = bv[dch];
    float aaL = aa;
#pragma unroll
    for (int i = 0; i < 6; ++i) aaL *= aaL;            // aa^64 (LCH = 2^6)

    float s = 0.f, msum = 0.f;
#pragma unroll 1
    for (int c = 0; c < NCHUNK; ++c) {
        init[c * BD + bd] = s;
        s = fmaf(aaL, s, bb * e_part[c * BD + bd]);
        msum += sum_part[c * BD + bd];
    }
    u_mean[bd] = msum * (1.0f / (float)TT);
}

// ---------------------------------------------------------------------------
// Kernel 4: d_eff[b][d] = d[d] + ((u_mean[b] @ d_lr_u) @ d_lr_v)[d] / R
// ---------------------------------------------------------------------------
__global__ __launch_bounds__(256)
void ssm_lowrank_kernel(const float* __restrict__ u_mean,
                        const float* __restrict__ dvec,
                        const float* __restrict__ dlr_u,
                        const float* __restrict__ dlr_v,
                        float* __restrict__ d_eff) {
    __shared__ float red[256];
    __shared__ float tmp[BB * RR];
    const int tid = threadIdx.x;

    for (int b = 0; b < BB; ++b) {
        for (int r = 0; r < RR; ++r) {
            float s = 0.f;
            for (int dd = tid; dd < DD; dd += 256)
                s += u_mean[b * DD + dd] * dlr_u[dd * RR + r];
            red[tid] = s;
            __syncthreads();
            for (int off = 128; off > 0; off >>= 1) {
                if (tid < off) red[tid] += red[tid + off];
                __syncthreads();
            }
            if (tid == 0) tmp[b * RR + r] = red[0];
            __syncthreads();
        }
    }
    for (int i = tid; i < BD; i += 256) {
        int b   = i >> 11;
        int dch = i & (DD - 1);
        float a = 0.f;
#pragma unroll
        for (int r = 0; r < RR; ++r)
            a += tmp[b * RR + r] * dlr_v[r * DD + dch];
        d_eff[i] = dvec[dch] + a * (1.0f / (float)RR);
    }
}

// ---------------------------------------------------------------------------
// Kernel 5: chunked rescan, in place over d_out (reads u_g, writes y).
//   s = init[c];  s[t] = aa*s + b*x;  y[t] = c*s + d_eff*x
// One thread per (chunk, b, d).
// ---------------------------------------------------------------------------
__global__ __launch_bounds__(256)
void ssm_rescan_kernel(const float* __restrict__ av,
                       const float* __restrict__ bv,
                       const float* __restrict__ cv,
                       const float* __restrict__ d_eff,
                       const float* __restrict__ init,
                       float* __restrict__ io) {
    int idx = blockIdx.x * blockDim.x + threadIdx.x;
    int bd  = idx & (BD - 1);
    int c   = idx >> BD_SHIFT;
    int dch = bd & (DD - 1);
    int b   = bd >> 11;

    float aa = tanhf(av[dch]);
    float bb = bv[dch];
    float cc = cv[dch];
    float de = d_eff[bd];
    float s  = init[c * BD + bd];

    float* p = io + ((size_t)b * TT + (size_t)c * LCH) * DD + dch;
#pragma unroll 4
    for (int i = 0; i < LCH; ++i) {
        float x = p[(size_t)i * DD];
        s = fmaf(aa, s, bb * x);
        p[(size_t)i * DD] = fmaf(cc, s, de * x);
    }
}

// ---------------------------------------------------------------------------
// Launcher
// ---------------------------------------------------------------------------
extern "C" void kernel_launch(void* const* d_in, const int* in_sizes, int n_in,
                              void* d_out, int out_size, void* d_ws, size_t ws_size,
                              hipStream_t stream) {
    (void)in_sizes; (void)n_in; (void)out_size;
    const float* u      = (const float*)d_in[0];
    const float* a      = (const float*)d_in[1];
    const float* b      = (const float*)d_in[2];
    const float* c      = (const float*)d_in[3];
    const float* d      = (const float*)d_in[4];
    const float* gate_w = (const float*)d_in[5];
    const float* gate_b = (const float*)d_in[6];
    const float* dlr_u  = (const float*)d_in[7];
    const float* dlr_v  = (const float*)d_in[8];
    float* out = (float*)d_out;

    const size_t u_elems = (size_t)BB * TT * DD;
    const size_t w_elems = (size_t)DD * DD;
    const size_t scan_floats = 3 * (size_t)NCHUNK * BD + 2 * (size_t)BD;
    const size_t need_big = u_elems * 2 + w_elems * 2 + scan_floats * 4;

    const int M = BB * TT;
    dim3 gemm_grid((M / TILE_M) * (DD / TILE_N));

    float* scan_base;
    if (ws_size >= need_big) {
        // Primary path: bf16 pre-convert + async double-buffered WMMA GEMM.
        __bf16* u_bf = (__bf16*)d_ws;
        __bf16* w_bf = u_bf + u_elems;
        scan_base = (float*)(w_bf + w_elems);

        cvt_bf16_kernel<<<dim3(4096), 256, 0, stream>>>(u, u_bf, (int)(u_elems / 4));
        cvt_bf16_kernel<<<dim3(512),  256, 0, stream>>>(gate_w, w_bf, (int)(w_elems / 4));
        ssm_gate_gemm_async_kernel<<<gemm_grid, 256, 0, stream>>>(u_bf, w_bf, u,
                                                                  gate_b, out);
    } else {
        // Fallback: convert-in-kernel GEMM, small ws (~6.2 MB).
        scan_base = (float*)d_ws;
        ssm_gate_gemm_kernel<<<gemm_grid, 256, 0, stream>>>(u, gate_w, gate_b, out);
    }

    float* e_part   = scan_base;
    float* sum_part = e_part + (size_t)NCHUNK * BD;
    float* init     = sum_part + (size_t)NCHUNK * BD;
    float* u_mean   = init + (size_t)NCHUNK * BD;
    float* d_eff    = u_mean + BD;

    dim3 chunk_grid((BD * NCHUNK) / 256);   // 2048 blocks, 16K waves
    ssm_chunk_reduce_kernel<<<chunk_grid, 256, 0, stream>>>(out, a, e_part, sum_part);
    ssm_combine_kernel<<<dim3(BD / 256), 256, 0, stream>>>(e_part, sum_part, a, b,
                                                           init, u_mean);
    ssm_lowrank_kernel<<<dim3(1), 256, 0, stream>>>(u_mean, d, dlr_u, dlr_v, d_eff);
    ssm_rescan_kernel<<<chunk_grid, 256, 0, stream>>>(a, b, c, d_eff, init, out);
}
